// RGCNNaivePsychology_8693013807318
// MI455X (gfx1250) — matrix operand order repaired and checked by reading, not compiled, not run on hardware
//
#include <hip/hip_runtime.h>

// ---------------------------------------------------------------------------
// RGCN (scatter-then-GEMM formulation) + MLP head, BF16 WMMA on gfx1250.
//   agg = Sum_r segment_sum(norm*h[src] | type==r) @ W_r  +  h @ W_loop
//   h'  = relu(agg + b)
// GEMMs: dense [M,K]x[K,N] bf16 WMMA, f32 accumulate.
// Block tile 128x64, 8 wave32s (4M x 2N), 32x32 per wave = 4 WMMA / k-step.
// Double-buffered LDS, ONE barrier per k-step:
//   A tile staged with GLOBAL_LOAD_ASYNC_TO_LDS_B128 (ASYNCcnt path),
//   B tile staged synchronously (transposed on the way into LDS),
//   both one iteration ahead, overlapping the WMMAs.
// ---------------------------------------------------------------------------

#define Nn   20000
#define Dd   768
#define Rr   8
#define Ee   100000
#define Ll   2
#define H1d  (Dd/2)   // 384
#define H2d  (Dd/4)   // 192
#define Cc   19

typedef __attribute__((ext_vector_type(16))) __bf16          v16bf;
typedef __attribute__((ext_vector_type(16))) unsigned short  v16u;
typedef __attribute__((ext_vector_type(8)))  unsigned short  ushort8;
typedef __attribute__((ext_vector_type(8)))  float           v8f;

__device__ __forceinline__ unsigned short f32_to_bf16(float f) {
    unsigned int u = __builtin_bit_cast(unsigned int, f);
    u += 0x7FFFu + ((u >> 16) & 1u);            // round-to-nearest-even
    return (unsigned short)(u >> 16);
}

// async 16B global -> LDS copy (per-lane), tracked by ASYNCcnt
__device__ __forceinline__ void async_copy_b128(unsigned short* lds_dst,
                                                const unsigned short* gsrc) {
    unsigned lds_off = (unsigned)(unsigned long long)lds_dst; // low 32b = LDS offset
    asm volatile("global_load_async_to_lds_b128 %0, %1, off"
                 :: "v"(lds_off), "v"(gsrc) : "memory");
}
__device__ __forceinline__ void wait_asynccnt0() {
    asm volatile("s_wait_asynccnt 0x0" ::: "memory");
}

// -------------------------- elementwise kernels ----------------------------

__global__ __launch_bounds__(256) void cvt_f32_bf16_kernel(
    const float* __restrict__ src, unsigned short* __restrict__ dst, long n) {
    long i = (long)blockIdx.x * blockDim.x + threadIdx.x;
    if (i < n) dst[i] = f32_to_bf16(src[i]);
}

__global__ __launch_bounds__(256) void bias_relu_bf16_kernel(
    const float* __restrict__ agg, const float* __restrict__ bias,
    float* __restrict__ h, unsigned short* __restrict__ hb, long n, int D) {
    long i = (long)blockIdx.x * blockDim.x + threadIdx.x;
    if (i >= n) return;
    int col = (int)(i % D);
    float v = agg[i] + bias[col];
    v = v > 0.f ? v : 0.f;
    h[i]  = v;
    hb[i] = f32_to_bf16(v);
}

// S[dst,:] += norm_e * h[src,:] for edges of relation `rel`
__global__ __launch_bounds__(256) void scatter_rel_kernel(
    const float* __restrict__ h, const float* __restrict__ norm,
    const int* __restrict__ esrc, const int* __restrict__ edst,
    const int* __restrict__ etype, int rel, float* __restrict__ S, int D) {
    int e = blockIdx.x;
    if (etype[e] != rel) return;
    float w = norm[e];
    const float* hs = h + (size_t)esrc[e] * D;
    float*       sd = S + (size_t)edst[e] * D;
    for (int j = threadIdx.x; j < D; j += blockDim.x)
        unsafeAtomicAdd(&sd[j], w * hs[j]);   // native global_atomic_add_f32
}

// ------------------------------ WMMA GEMM ----------------------------------
// Out-of-range rows/cols are CLAMPED (not predicated) during staging; garbage
// only reaches output tiles that the epilogue skips wholesale (M is a multiple
// of 16, so every 16-row wave subtile is fully in or fully out of range).
// K must be a multiple of 32 (768/384/192 all qualify).
// NFULL=true  : Nc is a multiple of 64, no column checks at all.

#define AST 40   // padded LDS row stride (elements); 80B rows keep 16B alignment

template<bool ACCUM, bool BIAS, bool RELU, bool WF32, bool WBF16, bool NFULL>
__global__ __launch_bounds__(256) void gemm_bf16_wmma_kernel(
    const unsigned short* __restrict__ A, const unsigned short* __restrict__ B,
    const float* __restrict__ bias, float* __restrict__ Cf,
    unsigned short* __restrict__ Cb, int M, int K, int Nc)
{
    __shared__ __align__(16) unsigned short As[2][128 * AST];
    __shared__ __align__(16) unsigned short Bs[2][64 * AST];  // transposed: Bs[p][col][k]

    const int tid  = threadIdx.x;
    const int wave = tid >> 5, lane = tid & 31;
    const int wm   = wave >> 1, wn = wave & 1;
    const int ln   = lane & 15, hi = lane >> 4;

    const int row0 = blockIdx.x * 128;
    const int n0   = blockIdx.y * 64;

    // staging assignments
    const int ar0 = tid >> 2;            // 0..63   (A rows, 2 passes of 64)
    const int ac  = (tid & 3) * 8;       // 0,8,16,24
    const int bk  = tid >> 3;            // 0..31   (B k-row in tile)
    const int bc  = (tid & 7) * 8;       // 0..56   (B col group of 8)

    // clamped global A rows for this thread (address-safe; garbage discarded)
    int grow0 = row0 + ar0;       if (grow0 > M - 1) grow0 = M - 1;
    int grow1 = row0 + ar0 + 64;  if (grow1 > M - 1) grow1 = M - 1;

    auto stageA = [&](int p, int kk) {
        async_copy_b128(&As[p][ar0 * AST + ac],        A + (size_t)grow0 * K + kk + ac);
        async_copy_b128(&As[p][(ar0 + 64) * AST + ac], A + (size_t)grow1 * K + kk + ac);
    };
    auto stageB = [&](int p, int kk) {
        if (NFULL) {
            ushort8 bv = *(const ushort8*)(B + (size_t)(kk + bk) * Nc + n0 + bc);
            #pragma unroll
            for (int j = 0; j < 8; ++j) Bs[p][(bc + j) * AST + bk] = bv[j];
        } else {
            #pragma unroll
            for (int j = 0; j < 8; ++j) {
                int col = n0 + bc + j; if (col > Nc - 1) col = Nc - 1;   // clamp
                Bs[p][(bc + j) * AST + bk] = B[(size_t)(kk + bk) * Nc + col];
            }
        }
    };

    v8f acc[2][2] = {};

    // prologue: stage k0=0 into buffer 0
    stageA(0, 0);
    stageB(0, 0);

    int p = 0;
    for (int k0 = 0; k0 < K; k0 += 32, p ^= 1) {
        wait_asynccnt0();        // async A copies for buffer p complete
        __syncthreads();         // everyone done staging p / done reading p^1

        // stage next k-slice into the other buffer (overlaps the WMMAs below;
        // last iteration restages a clamped slice that is never read)
        const int kn = (k0 + 32 < K) ? k0 + 32 : k0;
        stageA(p ^ 1, kn);
        stageB(p ^ 1, kn);

        // ---- fragments per ISA wave32 layouts, then 4 WMMAs (2x2 reuse)
        union Frag { v16u u; ushort8 h[2]; v16bf b; } fa[2], fb[2];
        #pragma unroll
        for (int i = 0; i < 2; ++i) {
            const int arow = wm * 32 + i * 16 + ln;
            fa[i].h[0] = *(const ushort8*)&As[p][arow * AST + hi * 8];       // K kb..kb+7
            fa[i].h[1] = *(const ushort8*)&As[p][arow * AST + hi * 8 + 16];  // K kb+16..+23
        }
        #pragma unroll
        for (int j = 0; j < 2; ++j) {
            const int bcol = wn * 32 + j * 16 + ln;
            fb[j].h[0] = *(const ushort8*)&Bs[p][bcol * AST + hi * 8];       // K kb..kb+7
            fb[j].h[1] = *(const ushort8*)&Bs[p][bcol * AST + hi * 8 + 16];  // K kb+16..+23
        }
        #pragma unroll
        for (int i = 0; i < 2; ++i)
            #pragma unroll
            for (int j = 0; j < 2; ++j)
                acc[i][j] = __builtin_amdgcn_wmma_f32_16x16x32_bf16(
                    false, fa[i].b, false, fb[j].b, (short)0, acc[i][j], false, false);
    }

    // ---- epilogue: C/D layout — lane group hi selects M base (0 / 8).
    // M % 16 == 0 for all uses, so each 16-row wave subtile is all-or-nothing:
    // one wave-uniform branch, unguarded stores inside.
    #pragma unroll
    for (int j = 0; j < 2; ++j) {
        const int col = n0 + wn * 32 + j * 16 + ln;
        if (NFULL || col < Nc) {
            #pragma unroll
            for (int i = 0; i < 2; ++i) {
                const int rt = row0 + wm * 32 + i * 16;     // wave-uniform
                if (rt + 16 <= M) {
                    const int rbase = rt + hi * 8;
                    #pragma unroll
                    for (int v = 0; v < 8; ++v) {
                        float val = acc[i][j][v];
                        size_t idx = (size_t)(rbase + v) * Nc + col;
                        if (ACCUM) val += Cf[idx];
                        if (BIAS)  val += bias[col];
                        if (RELU)  val = val > 0.f ? val : 0.f;
                        if (WF32)  Cf[idx] = val;
                        if (WBF16) Cb[idx] = f32_to_bf16(val);
                    }
                }
            }
        }
    }
}

// ------------------------------- launcher ----------------------------------

static inline void cvt(const float* s, unsigned short* d, long n, hipStream_t st) {
    cvt_f32_bf16_kernel<<<(unsigned)((n + 255) / 256), 256, 0, st>>>(s, d, n);
}

extern "C" void kernel_launch(void* const* d_in, const int* in_sizes, int n_in,
                              void* d_out, int out_size, void* d_ws, size_t ws_size,
                              hipStream_t stream) {
    (void)in_sizes; (void)n_in; (void)out_size; (void)ws_size;

    const float* node_feat = (const float*)d_in[0];
    const float* edge_norm = (const float*)d_in[1];
    const float* W_rel     = (const float*)d_in[2];   // [L,R,D,D]
    const float* W_loop    = (const float*)d_in[3];   // [L,D,D]
    const float* b_rgcn    = (const float*)d_in[4];   // [L,D]
    const float* l1_w      = (const float*)d_in[5];   // [D,H1]
    const float* l1_b      = (const float*)d_in[6];
    const float* l2_w      = (const float*)d_in[7];   // [H1,H2]
    const float* l2_b      = (const float*)d_in[8];
    const float* lout_w    = (const float*)d_in[9];   // [H2,C]
    const float* lout_b    = (const float*)d_in[10];
    const int*   edge_src  = (const int*)d_in[11];
    const int*   edge_dst  = (const int*)d_in[12];
    const int*   edge_type = (const int*)d_in[13];
    float*       out       = (float*)d_out;

    // ---- carve workspace
    size_t off = 0;
    char* base = (char*)d_ws;
    auto carve = [&](size_t bytes) -> void* {
        void* p = base + off;
        off += (bytes + 255) & ~(size_t)255;
        return p;
    };
    float*          h_buf  = (float*)carve((size_t)Nn * Dd * 4);
    unsigned short* hb     = (unsigned short*)carve((size_t)Nn * Dd * 2);
    float*          agg    = (float*)carve((size_t)Nn * Dd * 4);
    float*          S      = (float*)carve((size_t)Nn * Dd * 4);
    unsigned short* Sb     = (unsigned short*)carve((size_t)Nn * Dd * 2);
    unsigned short* wrelb  = (unsigned short*)carve((size_t)Ll * Rr * Dd * Dd * 2);
    unsigned short* wloopb = (unsigned short*)carve((size_t)Ll * Dd * Dd * 2);
    unsigned short* l1b    = (unsigned short*)carve((size_t)Dd * H1d * 2);
    unsigned short* l2b    = (unsigned short*)carve((size_t)H1d * H2d * 2);
    unsigned short* loutb  = (unsigned short*)carve((size_t)H2d * Cc * 2);

    // ---- weights -> bf16 (every call; deterministic)
    cvt(W_rel,  wrelb,  (long)Ll * Rr * Dd * Dd, stream);
    cvt(W_loop, wloopb, (long)Ll * Dd * Dd,      stream);
    cvt(l1_w,   l1b,    (long)Dd * H1d,          stream);
    cvt(l2_w,   l2b,    (long)H1d * H2d,         stream);
    cvt(lout_w, loutb,  (long)H2d * Cc,          stream);
    cvt(node_feat, hb,  (long)Nn * Dd,           stream);

    const dim3 gD((Nn + 127) / 128, Dd / 64);

    const float* hcur = node_feat;
    for (int l = 0; l < Ll; ++l) {
        // agg = h @ W_loop[l]   (non-accumulating first pass)
        gemm_bf16_wmma_kernel<false,false,false,true,false,true><<<gD, 256, 0, stream>>>(
            hb, wloopb + (size_t)l * Dd * Dd, nullptr, agg, nullptr, Nn, Dd, Dd);

        for (int r = 0; r < Rr; ++r) {
            hipMemsetAsync(S, 0, (size_t)Nn * Dd * 4, stream);
            scatter_rel_kernel<<<Ee, 256, 0, stream>>>(
                hcur, edge_norm, edge_src, edge_dst, edge_type, r, S, Dd);
            cvt(S, Sb, (long)Nn * Dd, stream);
            // agg += S_r @ W_rel[l][r]
            gemm_bf16_wmma_kernel<true,false,false,true,false,true><<<gD, 256, 0, stream>>>(
                Sb, wrelb + ((size_t)l * Rr + r) * Dd * Dd, nullptr, agg, nullptr,
                Nn, Dd, Dd);
        }
        // h = relu(agg + b);  also emit bf16 copy for the next GEMM stage
        bias_relu_bf16_kernel<<<(unsigned)(((long)Nn * Dd + 255) / 256), 256, 0, stream>>>(
            agg, b_rgcn + (size_t)l * Dd, h_buf, hb, (long)Nn * Dd, Dd);
        hcur = h_buf;
    }

    // ---- MLP head: h1 = relu(h@l1_w+b) -> Sb ; h2 = relu(h1@l2_w+b) -> hb ;
    //      out = h2@lout_w + b  (f32)
    dim3 g1((Nn + 127) / 128, H1d / 64);
    gemm_bf16_wmma_kernel<false,true,true,false,true,true><<<g1, 256, 0, stream>>>(
        hb, l1b, l1_b, nullptr, Sb, Nn, Dd, H1d);

    dim3 g2((Nn + 127) / 128, H2d / 64);
    gemm_bf16_wmma_kernel<false,true,true,false,true,true><<<g2, 256, 0, stream>>>(
        Sb, l2b, l2_b, nullptr, hb, Nn, H1d, H2d);

    dim3 g3((Nn + 127) / 128, (Cc + 63) / 64);
    gemm_bf16_wmma_kernel<false,true,false,true,false,false><<<g3, 256, 0, stream>>>(
        hb, loutb, lout_b, out, nullptr, Nn, H2d, Cc);
}
